// Agent_39522289057990
// MI455X (gfx1250) — matrix operand order repaired and checked by reading, not compiled, not run on hardware
//
#include <hip/hip_runtime.h>
#include <hip/hip_bf16.h>

// ---------------------------------------------------------------------------
// CDNA5 (gfx1250) WMMA actor-critic forward.
// All GEMM-shaped work uses v_wmma_f32_16x16x32_bf16 (bf16 in, f32 accum).
// LSTM x-tiles are DMAed into LDS with the Tensor Data Mover (TENSORcnt).
// ---------------------------------------------------------------------------

typedef __attribute__((ext_vector_type(16))) __bf16 v16bf;
typedef __attribute__((ext_vector_type(8)))  float  v8f;
typedef __attribute__((ext_vector_type(4)))  unsigned int u32x4;
typedef __attribute__((ext_vector_type(8)))  int    i32x8;
typedef __attribute__((ext_vector_type(4)))  int    i32x4;

union ABf {
  v16bf          v;
  unsigned short u[16];
  uint4          q[2];   // two 16B chunks = 8 VGPRs
};

__device__ inline unsigned short f2bf(float x) {
  unsigned int u = __float_as_uint(x);
  unsigned int r = (u + 0x7FFFu + ((u >> 16) & 1u)) >> 16;  // RNE
  return (unsigned short)r;
}
__device__ inline float bf2f(unsigned short b) {
  return __uint_as_float(((unsigned int)b) << 16);
}
__device__ inline float ldval(const float* p, long i) { return p[i]; }
__device__ inline float ldval(const unsigned short* p, long i) { return bf2f(p[i]); }
__device__ inline v8f vzero() { v8f z; for (int i = 0; i < 8; i++) z[i] = 0.f; return z; }
__device__ inline float sigm(float x) { return 1.f / (1.f + expf(-x)); }

// ---------------------------------------------------------------------------
// fp32 -> bf16 weight conversion
// ---------------------------------------------------------------------------
__global__ void cvt_bf16_kernel(const float* __restrict__ s,
                                unsigned short* __restrict__ d, long n) {
  long i = (long)blockIdx.x * blockDim.x + threadIdx.x;
  long st = (long)gridDim.x * blockDim.x;
  for (; i < n; i += st) d[i] = f2bf(s[i]);
}

// ---------------------------------------------------------------------------
// Implicit-GEMM conv via WMMA.
//   GEMM view: M = N*OH*OW output pixels, N = COUT, K = CIN*KH*KW.
//   A layout (16-bit A 16x32): lane%16 = M row; elems 0..7 = K half*8..+7,
//   elems 8..15 = K 16+half*8..+7.
//   B layout (16-bit B 32x16): lane%16 = N col; elems 0..7 = K half*8..+7,
//   elems 8..15 = K 16+half*8..+7 (two 16B runs of the weight row).
// ---------------------------------------------------------------------------
template <int CIN, int KH, int KW, int STR, int IH, int IW, int OH, int OW,
          int COUT, typename TIN>
__global__ void conv_wmma_kernel(const TIN* __restrict__ in,
                                 const unsigned short* __restrict__ wbf,
                                 const float* __restrict__ bias,
                                 unsigned short* __restrict__ out) {
  constexpr int K   = CIN * KH * KW;   // multiple of 32 for all three convs
  constexpr int NT  = COUT / 16;
  constexpr int MPI = OH * OW;

  const int lane = threadIdx.x & 31;
  const int wave = threadIdx.x >> 5;
  const int half = lane >> 4;
  const int lm   = lane & 15;
  const long mtile = (long)blockIdx.x * 4 + wave;

  // A-row mapping for this lane (output pixel index -> n, oh, ow)
  const long mrow = mtile * 16 + lm;
  const int n  = (int)(mrow / MPI);
  const int rm = (int)(mrow % MPI);
  const int oh = rm / OW, ow = rm % OW;
  const long inbase = (long)n * CIN * IH * IW;

  v8f acc[NT];
  for (int t = 0; t < NT; t++) acc[t] = vzero();

  for (int kb = 0; kb < K; kb += 32) {
    ABf a;
    if constexpr (CIN == 1 && KH == 8 && KW == 8) {
      // conv1: each 8-element K-run is one contiguous row of x.
      // Row bases are 16B-aligned (236*4 and ow*4*4 are multiples of 16).
      const int khA = (kb >> 3) + half;       // elems 0..7
      const int khB = khA + 2;                // elems 8..15
      const float* p0 = (const float*)in + inbase +
                        (long)(oh * STR + khA) * IW + ow * STR;
      const float* p1 = (const float*)in + inbase +
                        (long)(oh * STR + khB) * IW + ow * STR;
      float4 r0 = *(const float4*)p0, r1 = *(const float4*)(p0 + 4);
      float4 r2 = *(const float4*)p1, r3 = *(const float4*)(p1 + 4);
      a.u[0]  = f2bf(r0.x); a.u[1]  = f2bf(r0.y); a.u[2]  = f2bf(r0.z); a.u[3]  = f2bf(r0.w);
      a.u[4]  = f2bf(r1.x); a.u[5]  = f2bf(r1.y); a.u[6]  = f2bf(r1.z); a.u[7]  = f2bf(r1.w);
      a.u[8]  = f2bf(r2.x); a.u[9]  = f2bf(r2.y); a.u[10] = f2bf(r2.z); a.u[11] = f2bf(r2.w);
      a.u[12] = f2bf(r3.x); a.u[13] = f2bf(r3.y); a.u[14] = f2bf(r3.z); a.u[15] = f2bf(r3.w);
    } else {
#pragma unroll
      for (int e = 0; e < 16; e++) {
        int k   = kb + ((e >> 3) << 4) + (half << 3) + (e & 7);
        int cin = k / (KH * KW);
        int r2  = k - cin * (KH * KW);
        int kh  = r2 / KW, kw = r2 - kh * KW;
        float v = ldval(in, inbase + ((long)cin * IH + (oh * STR + kh)) * IW +
                                 (ow * STR + kw));
        a.u[e] = f2bf(v);
      }
    }
#pragma unroll
    for (int t = 0; t < NT; t++) {
      ABf b;
      const unsigned short* bp = wbf + (long)(t * 16 + lm) * K + kb + (half << 3);
      b.q[0] = *(const uint4*)bp;
      b.q[1] = *(const uint4*)(bp + 16);
      acc[t] = __builtin_amdgcn_wmma_f32_16x16x32_bf16(
          false, a.v, false, b.v, (short)0, acc[t], false, false);
    }
  }

  // D layout: lane%16 = N col; row M = r + (lane/16)*8
#pragma unroll
  for (int t = 0; t < NT; t++) {
    float bv = bias[t * 16 + lm];
#pragma unroll
    for (int r = 0; r < 8; r++) {
      long mm = mtile * 16 + (half << 3) + r;
      int nn = (int)(mm / MPI);
      int rr = (int)(mm % MPI);
      int o_h = rr / OW, o_w = rr - o_h * OW;
      float v = acc[t][r] + bv;
      v = fmaxf(v, 0.f);
      out[(((long)nn * COUT + t * 16 + lm) * OH + o_h) * OW + o_w] = f2bf(v);
    }
  }
}

// ---------------------------------------------------------------------------
// FC: feat = relu(flat @ fc_w^T + b).  M=1024, N=512, K=46592.
// Both A and B rows are K-contiguous -> all-b128 fragment loads.
// ---------------------------------------------------------------------------
__global__ void fc_wmma_kernel(const unsigned short* __restrict__ a3,
                               const unsigned short* __restrict__ wfc,
                               const float* __restrict__ fcb,
                               unsigned short* __restrict__ feat) {
  constexpr long K = 46592;
  const int lane = threadIdx.x & 31, wave = threadIdx.x >> 5;
  const int half = lane >> 4, lm = lane & 15;
  const int gw = blockIdx.x * 4 + wave;
  const int mt = gw >> 5;  // 64 M tiles
  const int nt = gw & 31;  // 32 N tiles

  const unsigned short* arow = a3 + (long)(mt * 16 + lm) * K;
  const unsigned short* brow = wfc + (long)(nt * 16 + lm) * K;

  v8f acc = vzero();
  for (long kb = 0; kb < K; kb += 32) {
    ABf a, b;
    a.q[0] = *(const uint4*)(arow + kb + (half << 3));
    a.q[1] = *(const uint4*)(arow + kb + 16 + (half << 3));
    const unsigned short* bp = brow + kb + (half << 3);
    b.q[0] = *(const uint4*)bp;
    b.q[1] = *(const uint4*)(bp + 16);
    __builtin_prefetch(bp + 128, 0, 1);  // global_prefetch_b8 on the B stream
    acc = __builtin_amdgcn_wmma_f32_16x16x32_bf16(
        false, a.v, false, b.v, (short)0, acc, false, false);
  }
  int j = nt * 16 + lm;
  float bv = fcb[j];
#pragma unroll
  for (int r = 0; r < 8; r++) {
    float v = acc[r] + bv;
    v = fmaxf(v, 0.f);
    feat[(long)(mt * 16 + (half << 3) + r) * 512 + j] = f2bf(v);
  }
}

// ---------------------------------------------------------------------------
// LSTM: persistent single workgroup (32 waves). B=16 == WMMA M.
// h (bf16) and c (f32) live in LDS.  The x_t tile (16x512 bf16, 16 KB) is
// DMAed into LDS once per step by the Tensor Data Mover (wave 0 issues
// tensor_load_to_lds, waits on TENSORcnt, then a workgroup barrier publishes
// it) instead of every wave re-reading it from global.
// Wave w owns gate columns w*16..w*16+15 of i/f/g/o (tiles w, 32+w, 64+w,
// 96+w) so the cell update is entirely in-register.
// ---------------------------------------------------------------------------
__global__ __launch_bounds__(1024) void lstm_kernel(
    const unsigned short* __restrict__ feat,
    const unsigned short* __restrict__ wih,
    const unsigned short* __restrict__ whh,
    const float* __restrict__ b_ih, const float* __restrict__ b_hh,
    const float* __restrict__ h0, const float* __restrict__ c0,
    const float* __restrict__ dones, float* __restrict__ hidden,
    float* __restrict__ out) {
  __shared__ __align__(16) unsigned short h_s[16 * 512];   // 16 KB
  __shared__ __align__(16) float          c_s[16 * 512];   // 32 KB
  __shared__ __align__(16) unsigned short x_s[16 * 512];   // 16 KB (TDM dest)

  const int tid  = threadIdx.x;
  const int lane = tid & 31, wave = tid >> 5;
  const int half = lane >> 4, lm = lane & 15;

  for (int i = tid; i < 8192; i += 1024) {
    h_s[i] = f2bf(h0[i]);
    c_s[i] = c0[i];
  }
  __syncthreads();

  for (int t = 0; t < 64; t++) {
    // done-based state reset
    for (int i = tid; i < 8192; i += 1024) {
      float m = 1.f - dones[t * 16 + (i >> 9)];
      c_s[i] *= m;
      h_s[i] = f2bf(bf2f(h_s[i]) * m);
    }

    if (wave == 0) {
      // Tensor DMA descriptor (ISA 8.3/8.4): 2D tensor, 16x512 bf16 tile.
      // For LDS-aperture generic pointers the low 32 address bits are the
      // LDS byte address.
      unsigned long long ga =
          (unsigned long long)(const void*)(feat + (long)t * 16 * 512);
      unsigned int lds = (unsigned int)(unsigned long long)(const void*)x_s;
      u32x4 g0;
      g0[0] = 1u;                                  // count=1 (valid D#)
      g0[1] = lds;                                 // lds_addr
      g0[2] = (unsigned int)ga;                    // global_addr[31:0]
      g0[3] = (unsigned int)(ga >> 32) | (2u << 30);  // addr[56:32] | type=2
      i32x8 g1;
      g1[0] = (int)(1u << 16);       // data_size = 2 bytes
      g1[1] = (int)(512u << 16);     // tensor_dim0 = 512 (bits 63:48)
      g1[2] = (int)(1024u << 16);    // tensor_dim1 = 1024 rows (bits 95:80)
      g1[3] = (int)(512u << 16);     // tile_dim0 = 512 (bits 127:112)
      g1[4] = 16;                    // tile_dim1 = 16
      g1[5] = 512;                   // tensor_dim0_stride = 512
      g1[6] = 0;
      g1[7] = 0;
      i32x4 gz4 = {0, 0, 0, 0};
      i32x8 gz8 = {0, 0, 0, 0, 0, 0, 0, 0};
      // 6-arg form (clang-23 / therock-10.0 lane)
      __builtin_amdgcn_tensor_load_to_lds(g0, g1, gz4, gz4, gz8, 0);
      __builtin_amdgcn_s_wait_tensorcnt(0);
    }
    __syncthreads();   // masked state + x tile visible to all waves

    v8f acc[4];
    for (int g = 0; g < 4; g++) acc[g] = vzero();

    const unsigned short* xrow = x_s + lm * 512;
    const unsigned short* hrow = h_s + lm * 512;

    for (int g = 0; g < 4; g++) {
      const int ncol = g * 512 + wave * 16 + lm;  // gate column -> weight row
      const unsigned short* bI = wih + (long)ncol * 512;
      const unsigned short* bH = whh + (long)ncol * 512;
      // x_t @ W_ih^T  (A fragments from LDS -> ds_read_b128)
      for (int kb = 0; kb < 512; kb += 32) {
        ABf a, b;
        a.q[0] = *(const uint4*)(xrow + kb + (half << 3));
        a.q[1] = *(const uint4*)(xrow + kb + 16 + (half << 3));
        const unsigned short* bp = bI + kb + (half << 3);
        b.q[0] = *(const uint4*)bp;
        b.q[1] = *(const uint4*)(bp + 16);
        acc[g] = __builtin_amdgcn_wmma_f32_16x16x32_bf16(
            false, a.v, false, b.v, (short)0, acc[g], false, false);
      }
      // h_{t-1} @ W_hh^T  (A fragments from LDS)
      for (int kb = 0; kb < 512; kb += 32) {
        ABf a, b;
        a.q[0] = *(const uint4*)(hrow + kb + (half << 3));
        a.q[1] = *(const uint4*)(hrow + kb + 16 + (half << 3));
        const unsigned short* bp = bH + kb + (half << 3);
        b.q[0] = *(const uint4*)bp;
        b.q[1] = *(const uint4*)(bp + 16);
        acc[g] = __builtin_amdgcn_wmma_f32_16x16x32_bf16(
            false, a.v, false, b.v, (short)0, acc[g], false, false);
      }
      float bb = b_ih[ncol] + b_hh[ncol];
#pragma unroll
      for (int r = 0; r < 8; r++) acc[g][r] += bb;
    }
    __syncthreads();  // all h_s reads done before overwrite

    const int col = wave * 16 + lm;
#pragma unroll
    for (int r = 0; r < 8; r++) {
      const int b   = (half << 3) + r;
      const int idx = b * 512 + col;
      float iv = acc[0][r], fv = acc[1][r], gv = acc[2][r], ov = acc[3][r];
      float cn = sigm(fv) * c_s[idx] + sigm(iv) * tanhf(gv);
      float hn = sigm(ov) * tanhf(cn);
      c_s[idx] = cn;
      h_s[idx] = f2bf(hn);
      hidden[(long)(t * 16 + b) * 512 + col] = hn;
      if (t == 63) {            // hT, cT outputs
        out[4096 + idx]  = hn;
        out[12288 + idx] = cn;
      }
    }
    __syncthreads();
  }
}

// ---------------------------------------------------------------------------
// Actor/critic heads: wave per row, shuffle reduction, log-softmax on lane 0.
// out layout: [0,1024) actions, [1024,2048) log_prob, [2048,3072) entropy,
//             [3072,4096) critic, [4096,12288) hT, [12288,20480) cT.
// ---------------------------------------------------------------------------
__global__ void heads_kernel(const float* __restrict__ hidden,
                             const float* __restrict__ aw,
                             const float* __restrict__ ab,
                             const float* __restrict__ cw,
                             const float* __restrict__ cb,
                             const int* __restrict__ actions,
                             float* __restrict__ out) {
  const int lane = threadIdx.x & 31;
  const int row  = blockIdx.x * 8 + (threadIdx.x >> 5);
  float acc[7];
  for (int j = 0; j < 7; j++) acc[j] = 0.f;
  const float* hr = hidden + (long)row * 512;
  for (int k = lane; k < 512; k += 32) {
    float h = hr[k];
#pragma unroll
    for (int a = 0; a < 6; a++) acc[a] += h * aw[a * 512 + k];
    acc[6] += h * cw[k];
  }
#pragma unroll
  for (int off = 16; off > 0; off >>= 1)
    for (int j = 0; j < 7; j++) acc[j] += __shfl_xor(acc[j], off, 32);
  if (lane == 0) {
    float lg[6];
    float mx = -1e30f;
    for (int a = 0; a < 6; a++) { lg[a] = acc[a] + ab[a]; mx = fmaxf(mx, lg[a]); }
    float se = 0.f;
    for (int a = 0; a < 6; a++) se += expf(lg[a] - mx);
    float lse = mx + logf(se);
    float ent = 0.f;
    for (int a = 0; a < 6; a++) { float lp = lg[a] - lse; ent -= expf(lp) * lp; }
    int act = actions[row];
    out[row]        = (float)act;
    out[1024 + row] = lg[act] - lse;
    out[2048 + row] = ent;
    out[3072 + row] = acc[6] + cb[0];
  }
}

// ---------------------------------------------------------------------------
// Workspace layout (bytes, 256-aligned).  Total ~473 MiB.
// ---------------------------------------------------------------------------
static constexpr size_t OFF_A1   = 0;                         // 1024*32*62*58 bf16
static constexpr size_t OFF_A2   = OFF_A1 + 235667456;        // 1024*64*30*28 bf16
static constexpr size_t OFF_A3   = OFF_A2 + 110100480;        // 1024*46592 bf16
static constexpr size_t OFF_FEAT = OFF_A3 + 95420416;         // 1024*512 bf16
static constexpr size_t OFF_HID  = OFF_FEAT + 1048576;        // 1024*512 f32
static constexpr size_t OFF_W1   = OFF_HID + 2097152;         // 2048 bf16
static constexpr size_t OFF_W2   = OFF_W1 + 4096;             // 32768 bf16
static constexpr size_t OFF_W3   = OFF_W2 + 65536;            // 36864 bf16
static constexpr size_t OFF_WFC  = OFF_W3 + 73728;            // 512*46592 bf16
static constexpr size_t OFF_WIH  = OFF_WFC + 47710208;        // 2048*512 bf16
static constexpr size_t OFF_WHH  = OFF_WIH + 2097152;         // 2048*512 bf16

extern "C" void kernel_launch(void* const* d_in, const int* in_sizes, int n_in,
                              void* d_out, int out_size, void* d_ws, size_t ws_size,
                              hipStream_t stream) {
  const float* x        = (const float*)d_in[0];
  const float* h0       = (const float*)d_in[1];
  const float* c0       = (const float*)d_in[2];
  const float* dones    = (const float*)d_in[3];
  const int*   actions  = (const int*)d_in[4];
  const float* c1_w     = (const float*)d_in[5];
  const float* c1_b     = (const float*)d_in[6];
  const float* c2_w     = (const float*)d_in[7];
  const float* c2_b     = (const float*)d_in[8];
  const float* c3_w     = (const float*)d_in[9];
  const float* c3_b     = (const float*)d_in[10];
  const float* fc_w     = (const float*)d_in[11];
  const float* fc_b     = (const float*)d_in[12];
  const float* w_ih     = (const float*)d_in[13];
  const float* w_hh     = (const float*)d_in[14];
  const float* b_ih     = (const float*)d_in[15];
  const float* b_hh     = (const float*)d_in[16];
  const float* actor_w  = (const float*)d_in[17];
  const float* actor_b  = (const float*)d_in[18];
  const float* critic_w = (const float*)d_in[19];
  const float* critic_b = (const float*)d_in[20];

  char* ws = (char*)d_ws;
  unsigned short* a1    = (unsigned short*)(ws + OFF_A1);
  unsigned short* a2    = (unsigned short*)(ws + OFF_A2);
  unsigned short* a3    = (unsigned short*)(ws + OFF_A3);
  unsigned short* featb = (unsigned short*)(ws + OFF_FEAT);
  float*          hid   = (float*)(ws + OFF_HID);
  unsigned short* w1b   = (unsigned short*)(ws + OFF_W1);
  unsigned short* w2b   = (unsigned short*)(ws + OFF_W2);
  unsigned short* w3b   = (unsigned short*)(ws + OFF_W3);
  unsigned short* wfcb  = (unsigned short*)(ws + OFF_WFC);
  unsigned short* wihb  = (unsigned short*)(ws + OFF_WIH);
  unsigned short* whhb  = (unsigned short*)(ws + OFF_WHH);

  // weights -> bf16
  cvt_bf16_kernel<<<8, 256, 0, stream>>>(c1_w, w1b, 2048L);
  cvt_bf16_kernel<<<128, 256, 0, stream>>>(c2_w, w2b, 32768L);
  cvt_bf16_kernel<<<144, 256, 0, stream>>>(c3_w, w3b, 36864L);
  cvt_bf16_kernel<<<8192, 256, 0, stream>>>(fc_w, wfcb, 23855104L);
  cvt_bf16_kernel<<<1024, 256, 0, stream>>>(w_ih, wihb, 1048576L);
  cvt_bf16_kernel<<<1024, 256, 0, stream>>>(w_hh, whhb, 1048576L);

  // CNN backbone (implicit GEMM, WMMA bf16)
  conv_wmma_kernel<1, 8, 8, 4, 252, 236, 62, 58, 32, float>
      <<<57536, 128, 0, stream>>>(x, w1b, c1_b, a1);
  conv_wmma_kernel<32, 4, 4, 2, 62, 58, 30, 28, 64, unsigned short>
      <<<13440, 128, 0, stream>>>(a1, w2b, c2_b, a2);
  conv_wmma_kernel<64, 3, 3, 1, 30, 28, 28, 26, 64, unsigned short>
      <<<11648, 128, 0, stream>>>(a2, w3b, c3_b, a3);

  // FC 1024x512x46592
  fc_wmma_kernel<<<512, 128, 0, stream>>>(a3, wfcb, fc_b, featb);

  // Recurrent core (persistent workgroup, TDM-fed x tiles)
  lstm_kernel<<<1, 1024, 0, stream>>>(featb, wihb, whhb, b_ih, b_hh, h0, c0,
                                      dones, hid, (float*)d_out);

  // Heads
  heads_kernel<<<128, 256, 0, stream>>>(hid, actor_w, actor_b, critic_w,
                                        critic_b, actions, (float*)d_out);
}
